// Observer_66872640799183
// MI455X (gfx1250) — matrix operand order repaired and checked by reading, not compiled, not run on hardware
//
#include <hip/hip_runtime.h>
#include <hip/hip_bf16.h>
#include <math.h>

// ---------------------------------------------------------------------------
// Batched TreeLSTM (NaryCell + ChildSumCell) for MI455X (gfx1250, wave32).
// bf16 WMMA (v_wmma_f32_16x16x32_bf16) with f32 accumulation.
// Nary: 64-row M-tile (4 m-tiles/wave) => ~32 WMMA-FLOP per L2 byte of
// streamed B; K-split LDS staging (2 x 384) stays under 64KB static LDS;
// gate-aligned wave->column-tile assignment keeps the sigmoid/tanh epilogue
// entirely in registers. B fragments are double-buffered in registers so one
// L2 load is always in flight beneath the WMMA stream.
// ---------------------------------------------------------------------------

typedef __attribute__((ext_vector_type(16))) __bf16 v16bf;
typedef __attribute__((ext_vector_type(8)))  __bf16 v8bf;
typedef __attribute__((ext_vector_type(8)))  float  v8f;

#define MQ   65536   // Nary nodes
#define M2Q  32768   // ChildSum nodes
#define HQ   256

// swizzled weight fragment counts (each fragment = 32 lanes x 16 bf16 = 512)
#define BN_KT 24
#define BN_NT 80
#define NBN  (BN_KT * BN_NT * 512)   // Nary combined B: [768 x 1280]
#define B1_KT 16
#define B1_NT 64
#define NB1  (B1_KT * B1_NT * 512)   // ChildSum GEMM1 B: [512 x 1024]
#define B2_KT 8
#define B2_NT 16
#define NB2  (B2_KT * B2_NT * 512)   // ChildSum GEMM2 B: [256 x 256]

// LDS strides (elems), padded to spread banks, rows 16B aligned
#define AP_STR  392   // Nary K-split A tile: 384 + 8 (bf16)
#define AX_STR  520   // ChildSum (x|hsum) rows: 512 + 8 (bf16)
#define AH_STR  264   // ChildSum per-child h rows: 256 + 8 (bf16)

__device__ __forceinline__ float sigm(float x) { return 1.0f / (1.0f + __expf(-x)); }

union FragA { v16bf v; struct { v8bf lo, hi; } s; };

__device__ __forceinline__ v16bf loadBfrag(const __bf16* __restrict__ Bsw,
                                           int frag, int lane)
{
    return *(const v16bf*)&Bsw[(size_t)frag * 512 + lane * 16];
}

// ---------------------------------------------------------------------------
// Weight prep: build bf16 fragment-swizzled B matrices into workspace.
// Fragment (kt,nt): element [lane*16 + e] = B[kt*32 + (lane>=16?8:0) + (e&7) + (e>=8?16:0)]
//                                            [nt*16 + (lane&15)]
// ---------------------------------------------------------------------------
__global__ void prep_weights(const float* __restrict__ U,
                             const float* __restrict__ W,
                             const float* __restrict__ Wf,
                             const float* __restrict__ Ucw,
                             const float* __restrict__ Wcw,
                             __bf16* __restrict__ ws)
{
    int idx = blockIdx.x * 256 + threadIdx.x;
    if (idx >= NBN + NB1 + NB2) return;

    float val = 0.0f;
    if (idx < NBN) {
        int e = idx & 15, lane = (idx >> 4) & 31, frag = idx >> 9;
        int kt = frag / BN_NT, nt = frag % BN_NT;
        int kk = kt * 32 + ((lane >> 4) << 3) + (e & 7) + ((e >> 3) << 4);
        int nn = nt * 16 + (lane & 15);
        // Bn[768 x 1280]: rows 0:256 = x (U), 256:512 = h0 (W0|Wf0), 512:768 = h1
        // cols 0:768 = i,j,o ; 768:1024 = f0 ; 1024:1280 = f1
        if (kk < 256) {
            int col = (nn < 1024) ? nn : (nn - 256);   // U's f-block feeds both f0,f1
            val = U[kk * 1024 + col];
        } else {
            int ch = (kk - 256) >> 8;
            int r  = (kk - 256) & 255;
            if (nn < 768) val = W[(ch * 256 + r) * 768 + nn];
            else          val = Wf[(ch * 256 + r) * 512 + (nn - 768)];
        }
    } else if (idx < NBN + NB1) {
        int j = idx - NBN;
        int e = j & 15, lane = (j >> 4) & 31, frag = j >> 9;
        int kt = frag >> 6, nt = frag & 63;
        int kk = kt * 32 + ((lane >> 4) << 3) + (e & 7) + ((e >> 3) << 4);
        int nn = nt * 16 + (lane & 15);
        // B1[512 x 1024]: rows 0:256 = x (Uc_w^T, i/j/o scaled by K=4),
        //                 rows 256:512 = hsum (Wc_w^T for i/j/o, 0 for f)
        if (kk < 256) {
            float u = Ucw[nn * 256 + kk];
            val = (nn < 768) ? 4.0f * u : u;
        } else {
            int r = kk - 256;
            val = (nn < 768) ? Wcw[nn * 256 + r] : 0.0f;
        }
    } else {
        int j = idx - NBN - NB1;
        int e = j & 15, lane = (j >> 4) & 31, frag = j >> 9;
        int kt = frag >> 4, nt = frag & 15;
        int kk = kt * 32 + ((lane >> 4) << 3) + (e & 7) + ((e >> 3) << 4);
        int nn = nt * 16 + (lane & 15);
        val = Wcw[(768 + nn) * 256 + kk];              // Wcf^T [256 x 256]
    }
    ws[idx] = (__bf16)val;
}

// ---------------------------------------------------------------------------
// NaryCell: 64 nodes per workgroup (4 m-tiles), 512 threads = 16 waves.
// GEMM [64x768] x [768x1280]; wave w owns column tiles {w,w+16,...,w+64} ==
// gates (i,j,o,f0,f1) for columns [16w,16w+16) -> register epilogue.
// K staged into LDS in two 384-wide phases (50KB buffer).
// ---------------------------------------------------------------------------
__global__ void __launch_bounds__(512)
nary_cell(const int*   __restrict__ nidx,
          const float* __restrict__ hs,
          const float* __restrict__ cs,
          const float* __restrict__ embed,
          const float* __restrict__ bias,
          const __bf16* __restrict__ Bsw,
          float* __restrict__ out)
{
    __shared__ __bf16 Alds[64 * AP_STR];   // 50176 B

    const int tid = threadIdx.x;
    const int g0  = blockIdx.x * 64;

    const int wave  = tid >> 5;
    const int lane  = tid & 31;
    const int mrow  = lane & 15;
    const int kbase = (lane >> 4) * 8;     // A 16-bit layout: lane halves

    v8f acc[5][4];                          // [gate][m-tile]
    #pragma unroll
    for (int j = 0; j < 5; ++j)
        #pragma unroll
        for (int mt = 0; mt < 4; ++mt)
            #pragma unroll
            for (int i = 0; i < 8; ++i) acc[j][mt][i] = 0.0f;

    for (int p = 0; p < 2; ++p) {
        if (p) __syncthreads();             // all waves done reading phase-0 A
        // stage A columns [384p, 384p+384) for 64 rows: (embed[x] | h0 | h1)
        for (int q = tid; q < 64 * 384; q += 512) {
            int m = q / 384, c = q - m * 384;
            int g = g0 + m;
            int col = p * 384 + c;
            float v;
            if (col < 256)      v = embed[nidx[g] * 256 + col];
            else if (col < 512) v = hs[(g * 2 + 0) * 256 + (col - 256)];
            else                v = hs[(g * 2 + 1) * 256 + (col - 512)];
            Alds[m * AP_STR + c] = (__bf16)v;
        }
        __syncthreads();

        for (int ktl = 0; ktl < 12; ++ktl) {
            int ktg = p * 12 + ktl;
            int k0  = ktl * 32 + kbase;

            // issue first B load of this k-step before the LDS A loads
            v16bf bcur = loadBfrag(Bsw, ktg * BN_NT + wave, lane);

            FragA a[4];
            #pragma unroll
            for (int mt = 0; mt < 4; ++mt) {
                int arow = mt * 16 + mrow;
                a[mt].s.lo = *(const v8bf*)&Alds[arow * AP_STR + k0];
                a[mt].s.hi = *(const v8bf*)&Alds[arow * AP_STR + k0 + 16];
            }

            if (ktg + 1 < BN_KT)
                __builtin_prefetch(&Bsw[(size_t)((ktg + 1) * BN_NT + wave) * 512 + lane * 16], 0, 3);

            #pragma unroll
            for (int j = 0; j < 5; ++j) {
                v16bf bnext = bcur;
                if (j < 4)                  // keep next gate-tile's load in flight
                    bnext = loadBfrag(Bsw, ktg * BN_NT + (wave + (j + 1) * 16), lane);
                #pragma unroll
                for (int mt = 0; mt < 4; ++mt)
                    acc[j][mt] = __builtin_amdgcn_wmma_f32_16x16x32_bf16(
                        false, a[mt].v, false, bcur, (short)0, acc[j][mt], false, false);
                bcur = bnext;
            }
        }
    }

    // register-resident epilogue: acc[0..4] = i,j,o,f0,f1 for col c
    const int c  = wave * 16 + (lane & 15);
    const float bi = bias[c], bj = bias[256 + c], bo = bias[512 + c], bf_ = bias[768 + c];
    #pragma unroll
    for (int mt = 0; mt < 4; ++mt) {
        #pragma unroll
        for (int v = 0; v < 8; ++v) {
            int m = mt * 16 + ((lane < 16) ? v : (v + 8));
            int g = g0 + m;
            float iv = acc[0][mt][v] + bi;
            float jv = acc[1][mt][v] + bj;
            float ov = acc[2][mt][v] + bo;
            float f0 = acc[3][mt][v] + bf_;
            float f1 = acc[4][mt][v] + bf_;
            float c0 = cs[(g * 2 + 0) * 256 + c];
            float c1 = cs[(g * 2 + 1) * 256 + c];
            float nc = sigm(f0) * c0 + sigm(f1) * c1 + sigm(iv) * tanhf(jv);
            float nh = sigm(ov) * tanhf(nc);
            out[(size_t)g * HQ + c]        = nh;   // next_h
            out[((size_t)MQ + g) * HQ + c] = nc;   // next_c
        }
    }
}

// ---------------------------------------------------------------------------
// ChildSumCell: 16 nodes per workgroup, 512 threads = 16 waves.
// GEMM1 [16x512](x|hsum) x [512x1024] -> i,j,o sums + x's f-contribution,
//   wave w owns tiles {w,w+16,w+32,w+48} (register-resident).
// GEMM2 [64x256](per-child h) x Wcf^T[256x256] -> per-child f, via LDS
//   (aliases the dead A-staging buffers; exactly 64 KB).
// ---------------------------------------------------------------------------
__global__ void __launch_bounds__(512)
childsum_cell(const int*   __restrict__ nidx,
              const float* __restrict__ hs,
              const float* __restrict__ cs,
              const float* __restrict__ embed,
              const float* __restrict__ Ucb,
              const float* __restrict__ Wcb,
              const __bf16* __restrict__ B1sw,
              const __bf16* __restrict__ B2sw,
              float* __restrict__ out)
{
    __shared__ __align__(16) char smem[65536];
    __bf16* Axh  = (__bf16*)smem;                    // [16 x AX_STR]  16640 B
    __bf16* Ah   = (__bf16*)(smem + 16640);          // [64 x AH_STR]  33792 B
    float*  lin2 = (float*)smem;                     // [64 x 256]     65536 B (aliased later)

    const int tid = threadIdx.x;
    const int g0  = blockIdx.x * 16;

    // stage (x | hsum) and per-child h as bf16
    for (int p = tid; p < 16 * 256; p += 512) {
        int m = p >> 8, c = p & 255;
        int g = g0 + m;
        Axh[m * AX_STR + c] = (__bf16)embed[nidx[g] * 256 + c];
        float s = 0.0f;
        #pragma unroll
        for (int k = 0; k < 4; ++k) {
            float hv = hs[(g * 4 + k) * 256 + c];
            Ah[(m * 4 + k) * AH_STR + c] = (__bf16)hv;
            s += hv;
        }
        Axh[m * AX_STR + 256 + c] = (__bf16)s;
    }
    __syncthreads();

    const int wave  = tid >> 5;
    const int lane  = tid & 31;
    const int mrow  = lane & 15;
    const int kbase = (lane >> 4) * 8;

    // ---- GEMM1: i,j,o sums + xf, register-resident per wave ----
    v8f acc1[4];
    #pragma unroll
    for (int j = 0; j < 4; ++j)
        #pragma unroll
        for (int i = 0; i < 8; ++i) acc1[j][i] = 0.0f;

    for (int kt = 0; kt < B1_KT; ++kt) {
        int k0 = kt * 32 + kbase;
        v16bf bcur = loadBfrag(B1sw, kt * B1_NT + wave, lane);
        FragA a;
        a.s.lo = *(const v8bf*)&Axh[mrow * AX_STR + k0];
        a.s.hi = *(const v8bf*)&Axh[mrow * AX_STR + k0 + 16];
        if (kt + 1 < B1_KT)
            __builtin_prefetch(&B1sw[(size_t)((kt + 1) * B1_NT + wave) * 512 + lane * 16], 0, 3);
        #pragma unroll
        for (int j = 0; j < 4; ++j) {
            v16bf bnext = bcur;
            if (j < 3)
                bnext = loadBfrag(B1sw, kt * B1_NT + (wave + (j + 1) * 16), lane);
            acc1[j] = __builtin_amdgcn_wmma_f32_16x16x32_bf16(
                false, a.v, false, bcur, (short)0, acc1[j], false, false);
            bcur = bnext;
        }
    }

    // ---- GEMM2: per-child f gates; wave w does all 4 M-tiles at nt=w ----
    v8f acc2[4];
    #pragma unroll
    for (int j = 0; j < 4; ++j)
        #pragma unroll
        for (int i = 0; i < 8; ++i) acc2[j][i] = 0.0f;

    {
        v16bf bcur = loadBfrag(B2sw, 0 * B2_NT + wave, lane);
        for (int kt = 0; kt < B2_KT; ++kt) {
            int k0 = kt * 32 + kbase;
            v16bf bnext = bcur;
            if (kt + 1 < B2_KT)             // pipeline B across k-steps
                bnext = loadBfrag(B2sw, (kt + 1) * B2_NT + wave, lane);
            #pragma unroll
            for (int mt = 0; mt < 4; ++mt) {
                int arow = mt * 16 + mrow;
                FragA a;
                a.s.lo = *(const v8bf*)&Ah[arow * AH_STR + k0];
                a.s.hi = *(const v8bf*)&Ah[arow * AH_STR + k0 + 16];
                acc2[mt] = __builtin_amdgcn_wmma_f32_16x16x32_bf16(
                    false, a.v, false, bcur, (short)0, acc2[mt], false, false);
            }
            bcur = bnext;
        }
    }

    __syncthreads();   // A buffers dead; safe to overwrite with lin2

    #pragma unroll
    for (int mt = 0; mt < 4; ++mt) {
        int col = wave * 16 + (lane & 15);
        #pragma unroll
        for (int v = 0; v < 8; ++v) {
            int row = mt * 16 + ((lane < 16) ? v : (v + 8));
            lin2[row * 256 + col] = acc2[mt][v];
        }
    }
    __syncthreads();

    // ---- epilogue: column chunk [16w,16w+16), all 16 nodes ----
    const int c = wave * 16 + (lane & 15);
    const float bi = 4.0f * (Ucb[c]       + Wcb[c]);
    const float bj = 4.0f * (Ucb[256 + c] + Wcb[256 + c]);
    const float bo = 4.0f * (Ucb[512 + c] + Wcb[512 + c]);
    const float bf_ =        Ucb[768 + c] + Wcb[768 + c];
    #pragma unroll
    for (int v = 0; v < 8; ++v) {
        int m = (lane < 16) ? v : (v + 8);
        int g = g0 + m;
        float ai = sigm(acc1[0][v] + bi);
        float aj = tanhf(acc1[1][v] + bj);
        float ao = sigm(acc1[2][v] + bo);
        float xf = acc1[3][v];
        float nc = ai * aj;
        #pragma unroll
        for (int k = 0; k < 4; ++k) {
            float f = sigm(xf + lin2[(m * 4 + k) * 256 + c] + bf_);
            nc += f * cs[(g * 4 + k) * 256 + c];
        }
        float nh = ao * tanhf(nc);
        out[((size_t)(2 * MQ) + g) * HQ + c]       = nh;   // next_h2
        out[((size_t)(2 * MQ + M2Q) + g) * HQ + c] = nc;   // next_c2
    }
}

// ---------------------------------------------------------------------------
extern "C" void kernel_launch(void* const* d_in, const int* in_sizes, int n_in,
                              void* d_out, int out_size, void* d_ws, size_t ws_size,
                              hipStream_t stream)
{
    const int*   idxN = (const int*)  d_in[0];
    const float* hsN  = (const float*)d_in[1];
    const float* csN  = (const float*)d_in[2];
    const int*   idxC = (const int*)  d_in[3];
    const float* hsC  = (const float*)d_in[4];
    const float* csC  = (const float*)d_in[5];
    const float* emb  = (const float*)d_in[6];
    const float* U    = (const float*)d_in[7];
    const float* W    = (const float*)d_in[8];
    const float* Wf   = (const float*)d_in[9];
    const float* b    = (const float*)d_in[10];
    const float* Ucw  = (const float*)d_in[11];
    const float* Ucb  = (const float*)d_in[12];
    const float* Wcw  = (const float*)d_in[13];
    const float* Wcb  = (const float*)d_in[14];
    float*  out = (float*)d_out;
    __bf16* wsb = (__bf16*)d_ws;

    const int total = NBN + NB1 + NB2;
    prep_weights<<<(total + 255) / 256, 256, 0, stream>>>(U, W, Wf, Ucw, Wcw, wsb);
    nary_cell<<<MQ / 64, 512, 0, stream>>>(idxN, hsN, csN, emb, b, wsb, out);
    childsum_cell<<<M2Q / 16, 512, 0, stream>>>(idxC, hsC, csC, emb, Ucb, Wcb,
                                                wsb + NBN, wsb + NBN + NB1, out);
}
